// DITAttention_1898375545314
// MI455X (gfx1250) — compile-verified, matrix operand order checked by reading, not run on hardware
//
#include <hip/hip_runtime.h>

// DiT attention for MI455X (gfx1250, wave32).
// All matmuls via v_wmma_f32_16x16x32_bf16 (densest non-8-bit WMMA: 16384 FLOP/inst,
// 8x the fp32 16x16x4 path). fp32 kept for softmax / RMSNorm / accumulators.
// Workspace assumption: ~152 MB of d_ws (x_bf16 16MB, 4 weights 32MB, one f32 temp
// 34MB, q/k/v/attn bf16 64MB+). Harness scratch is assumed >= this.

#define H_NUM 16
#define H_DIM 128

typedef __attribute__((ext_vector_type(16))) __bf16        v16bf;
typedef __attribute__((ext_vector_type(8)))  float         v8f;
typedef __attribute__((ext_vector_type(4)))  unsigned int  u32x4;

union Frag {            // one 16x32 (or 32x16) bf16 WMMA operand: 8 VGPRs
  v16bf bf;
  u32x4 q[2];
};

__device__ inline unsigned short f32_to_bf16(float f) {
  union { float f; unsigned int u; } c; c.f = f;
  unsigned int u = c.u + 0x7FFFu + ((c.u >> 16) & 1u);   // round-to-nearest-even
  return (unsigned short)(u >> 16);
}

// ---------------------------------------------------------------- convert f32 -> bf16
__global__ void cvt_f32_bf16_kernel(const float* __restrict__ in,
                                    unsigned short* __restrict__ out, long n) {
  long i = (long)blockIdx.x * blockDim.x + threadIdx.x;
  long stride = (long)gridDim.x * blockDim.x;
  for (; i < n; i += stride) out[i] = f32_to_bf16(in[i]);
}

// ---------------------------------------------------------------- generic NT GEMM
// C[M,N] = A[M,K] * B[N,K]^T, A/B bf16 row-major, C fp32 or bf16.
// Block = 256 threads (8 waves), 128x128 tile. Wave (wr,wc): 2 row-tiles x 4 col-tiles.
// NT symmetry: lane n of the B operand holds row n of B along K with the same packing
// as the A operand, so both fragments use identical b128-pair loads.
__global__ __launch_bounds__(256) void gemm_nt_bf16_kernel(
    const unsigned short* __restrict__ A, const unsigned short* __restrict__ B,
    float* __restrict__ Cf, unsigned short* __restrict__ Ch,
    int M, int N, int Kd, int store_bf16)
{
  const int tid  = threadIdx.x;
  const int wave = tid >> 5, lane = tid & 31;
  const int lm = lane & 15, lh = lane >> 4;        // lane-half K offset (A layout)
  const int wr = wave & 3, wc = wave >> 2;
  const long arow = (long)blockIdx.x * 128 + wr * 32;
  const long brow = (long)blockIdx.y * 128 + wc * 64;

  v8f acc[2][4];
#pragma unroll
  for (int i = 0; i < 2; ++i)
#pragma unroll
    for (int j = 0; j < 4; ++j)
#pragma unroll
      for (int e = 0; e < 8; ++e) acc[i][j][e] = 0.f;

  const unsigned short* pa0 = A + (arow + lm) * (long)Kd + lh * 8;
  const unsigned short* pb0 = B + (brow + lm) * (long)Kd + lh * 8;

  for (int k0 = 0; k0 < Kd; k0 += 32) {
    Frag a[2], b[4];
#pragma unroll
    for (int i = 0; i < 2; ++i) {
      const unsigned short* p = pa0 + (long)i * 16 * Kd + k0;
      a[i].q[0] = *(const u32x4*)p;
      a[i].q[1] = *(const u32x4*)(p + 16);
      __builtin_prefetch(p + 32, 0, 0);            // global_prefetch next K-step
    }
#pragma unroll
    for (int j = 0; j < 4; ++j) {
      const unsigned short* p = pb0 + (long)j * 16 * Kd + k0;
      b[j].q[0] = *(const u32x4*)p;
      b[j].q[1] = *(const u32x4*)(p + 16);
      __builtin_prefetch(p + 32, 0, 0);
    }
#pragma unroll
    for (int i = 0; i < 2; ++i)
#pragma unroll
      for (int j = 0; j < 4; ++j)
        acc[i][j] = __builtin_amdgcn_wmma_f32_16x16x32_bf16(
            false, a[i].bf, false, b[j].bf, (short)0, acc[i][j], false, false);
  }

  // C layout: VGPR r -> row (r + 8*laneHalf), lanes 0..15 -> col lm
#pragma unroll
  for (int i = 0; i < 2; ++i)
#pragma unroll
    for (int r = 0; r < 8; ++r) {
      long row = arow + i * 16 + r + 8 * lh;
#pragma unroll
      for (int j = 0; j < 4; ++j) {
        long col = brow + j * 16 + lm;
        float v = acc[i][j][r];
        if (store_bf16) Ch[row * (long)N + col] = f32_to_bf16(v);
        else            Cf[row * (long)N + col] = v;
      }
    }
}

// ---------------------------------------------------------------- RMSNorm + RoPE
// One wave per (b,s,h) row of 128; lane handles d = {lane, lane+32, lane+64, lane+96}.
// rope pairs (i, i+64): out[i] = f[i,0,0]*t[i] + f[i,0,1]*t[i+64]
//                       out[64+i] = f[i,1,0]*t[i] + f[i,1,1]*t[i+64]
__global__ __launch_bounds__(256) void rmsnorm_rope_kernel(
    const float* __restrict__ in, const float* __restrict__ w,
    const float* __restrict__ rope, unsigned short* __restrict__ out, int S)
{
  const int row  = blockIdx.x * 8 + (threadIdx.x >> 5);   // (b*S + s)*H_NUM + h
  const int lane = threadIdx.x & 31;
  const int s = (row >> 4) % S;
  const float* xr = in + (long)row * H_DIM;

  float v0 = xr[lane], v1 = xr[lane + 32], v2 = xr[lane + 64], v3 = xr[lane + 96];
  float ss = v0 * v0 + v1 * v1 + v2 * v2 + v3 * v3;
#pragma unroll
  for (int m = 16; m >= 1; m >>= 1) ss += __shfl_xor(ss, m, 32);
  float r = rsqrtf(ss * (1.0f / H_DIM) + 1e-6f);
  v0 *= r * w[lane];      v1 *= r * w[lane + 32];
  v2 *= r * w[lane + 64]; v3 *= r * w[lane + 96];

  const float* f = rope + (long)s * 256;                  // [64][2][2]
  float o0 = f[lane * 4 + 0] * v0 + f[lane * 4 + 1] * v2;
  float o2 = f[lane * 4 + 2] * v0 + f[lane * 4 + 3] * v2;
  float o1 = f[(lane + 32) * 4 + 0] * v1 + f[(lane + 32) * 4 + 1] * v3;
  float o3 = f[(lane + 32) * 4 + 2] * v1 + f[(lane + 32) * 4 + 3] * v3;

  unsigned short* yr = out + (long)row * H_DIM;
  yr[lane]      = f32_to_bf16(o0);
  yr[lane + 32] = f32_to_bf16(o1);
  yr[lane + 64] = f32_to_bf16(o2);
  yr[lane + 96] = f32_to_bf16(o3);
}

// ---------------------------------------------------------------- flash attention
// Grid: (S/128, B*H_NUM). Block = 8 waves; wave owns 16 q rows, d=128 in registers.
// Key tiles of 64. V tile transposed into LDS (Vt[d][k]); P goes through per-wave
// LDS to convert WMMA C-layout -> A-layout.
#define VT_STRIDE 72   // 64 keys + 8 pad (144B rows: 16B-aligned, bank-staggered)

__global__ __launch_bounds__(256) void attn_kernel(
    const unsigned short* __restrict__ Q, const unsigned short* __restrict__ K,
    const unsigned short* __restrict__ V, unsigned short* __restrict__ O, int S)
{
  __shared__ __align__(16) unsigned short Vt[H_DIM * VT_STRIDE];      // 18 KB
  __shared__ __align__(16) unsigned short Pl[8 * 16 * VT_STRIDE];     // 18 KB

  const int tid  = threadIdx.x;
  const int wave = tid >> 5, lane = tid & 31;
  const int lm = lane & 15, lh = lane >> 4;
  const int bh = blockIdx.y;
  const int b  = bh >> 4, h = bh & 15;
  const int q0 = blockIdx.x * 128 + wave * 16;

  auto rowbase = [&](int s) -> long {
    return ((long)(b * S + s)) * (H_NUM * H_DIM) + (long)h * H_DIM;
  };

  // preload Q fragments: 4 K-steps over head_dim=128
  Frag qf[4];
#pragma unroll
  for (int kk = 0; kk < 4; ++kk) {
    const unsigned short* p = Q + rowbase(q0 + lm) + kk * 32 + lh * 8;
    qf[kk].q[0] = *(const u32x4*)p;
    qf[kk].q[1] = *(const u32x4*)(p + 16);
  }

  v8f o[8];
#pragma unroll
  for (int t = 0; t < 8; ++t)
#pragma unroll
    for (int e = 0; e < 8; ++e) o[t][e] = 0.f;
  float m_run[8], l_run[8];
#pragma unroll
  for (int r = 0; r < 8; ++r) { m_run[r] = -1e30f; l_run[r] = 0.f; }
  const float scale = 0.08838834764831845f;   // 1/sqrt(128)

  for (int kt = 0; kt < S; kt += 64) {
    // ---- stage V tile [64 keys][128 d] transposed into Vt[d][key]
    {
      const int key = tid >> 2;               // 0..63
      const int dc  = (tid & 3) * 32;         // 32 contiguous d per thread
      const unsigned short* vp = V + rowbase(kt + key) + dc;
#pragma unroll
      for (int u = 0; u < 4; ++u) {
        u32x4 raw = *(const u32x4*)(vp + u * 8);
        const unsigned short* rs = (const unsigned short*)&raw;
#pragma unroll
        for (int e = 0; e < 8; ++e)
          Vt[(dc + u * 8 + e) * VT_STRIDE + key] = rs[e];
      }
    }
    __syncthreads();

    // ---- scores S = Q K^T for 16 q x 64 keys
    v8f sc[4];
#pragma unroll
    for (int j = 0; j < 4; ++j)
#pragma unroll
      for (int e = 0; e < 8; ++e) sc[j][e] = 0.f;
#pragma unroll
    for (int kk = 0; kk < 4; ++kk)
#pragma unroll
      for (int j = 0; j < 4; ++j) {
        const unsigned short* p = K + rowbase(kt + j * 16 + lm) + kk * 32 + lh * 8;
        Frag kf; kf.q[0] = *(const u32x4*)p; kf.q[1] = *(const u32x4*)(p + 16);
        sc[j] = __builtin_amdgcn_wmma_f32_16x16x32_bf16(
            false, qf[kk].bf, false, kf.bf, (short)0, sc[j], false, false);
      }

    // ---- online softmax (rows r / r+8 live in lane halves; xor masks 8..1 stay in-half)
#pragma unroll
    for (int r = 0; r < 8; ++r) {
      float mx = -1e30f;
#pragma unroll
      for (int j = 0; j < 4; ++j) mx = fmaxf(mx, sc[j][r]);
#pragma unroll
      for (int m = 8; m >= 1; m >>= 1) mx = fmaxf(mx, __shfl_xor(mx, m, 32));
      float mnew  = fmaxf(m_run[r], mx * scale);
      float alpha = __expf(m_run[r] - mnew);
      float rsum  = 0.f;
#pragma unroll
      for (int j = 0; j < 4; ++j) {
        float p = __expf(sc[j][r] * scale - mnew);
        rsum += p;
        Pl[(wave * 16 + r + 8 * lh) * VT_STRIDE + j * 16 + lm] = f32_to_bf16(p);
      }
#pragma unroll
      for (int m = 8; m >= 1; m >>= 1) rsum += __shfl_xor(rsum, m, 32);
      l_run[r] = l_run[r] * alpha + rsum;
      m_run[r] = mnew;
#pragma unroll
      for (int t = 0; t < 8; ++t) o[t][r] *= alpha;
    }

    // ---- O += P V  (A = P from LDS, B = Vt rows, K-dim = 64 keys = 2 steps)
#pragma unroll
    for (int kk = 0; kk < 2; ++kk) {
      Frag pa;
      const unsigned short* pp = &Pl[(wave * 16 + lm) * VT_STRIDE + kk * 32 + lh * 8];
      pa.q[0] = *(const u32x4*)pp;
      pa.q[1] = *(const u32x4*)(pp + 16);
#pragma unroll
      for (int t = 0; t < 8; ++t) {
        const unsigned short* vp = &Vt[(t * 16 + lm) * VT_STRIDE + kk * 32 + lh * 8];
        Frag vf; vf.q[0] = *(const u32x4*)vp; vf.q[1] = *(const u32x4*)(vp + 16);
        o[t] = __builtin_amdgcn_wmma_f32_16x16x32_bf16(
            false, pa.bf, false, vf.bf, (short)0, o[t], false, false);
      }
    }
    __syncthreads();
  }

  // ---- finalize and store bf16 attention output [b,s,h*128+d]
#pragma unroll
  for (int r = 0; r < 8; ++r) {
    float inv = 1.f / l_run[r];
    long base = rowbase(q0 + r + 8 * lh);
#pragma unroll
    for (int t = 0; t < 8; ++t)
      O[base + t * 16 + lm] = f32_to_bf16(o[t][r] * inv);
  }
}

// ---------------------------------------------------------------- launch
extern "C" void kernel_launch(void* const* d_in, const int* in_sizes, int n_in,
                              void* d_out, int out_size, void* d_ws, size_t ws_size,
                              hipStream_t stream) {
  (void)in_sizes; (void)n_in; (void)out_size; (void)ws_size;

  const float* x    = (const float*)d_in[0];
  const float* rope = (const float*)d_in[1];
  const float* Wq   = (const float*)d_in[2];
  const float* Wk   = (const float*)d_in[3];
  const float* Wv   = (const float*)d_in[4];
  const float* Wo   = (const float*)d_in[5];
  const float* qnw  = (const float*)d_in[6];
  const float* knw  = (const float*)d_in[7];

  const int  B = 2, S = 2048, D = 2048;
  const long M = (long)B * S;        // 4096 token rows
  const int  N = D;                  // 2048 (inner == D here)

  char* ws = (char*)d_ws;
  size_t off = 0;
  auto take = [&](size_t bytes) -> char* {
    char* p = ws + off;
    off += (bytes + 255) & ~(size_t)255;
    return p;
  };

  unsigned short* xb  = (unsigned short*)take(M * D * 2);
  unsigned short* wqb = (unsigned short*)take((long)D * D * 2);
  unsigned short* wkb = (unsigned short*)take((long)D * D * 2);
  unsigned short* wvb = (unsigned short*)take((long)D * D * 2);
  unsigned short* wob = (unsigned short*)take((long)D * D * 2);
  float*          tf  = (float*)take(M * N * 4);      // f32 temp, reused for q then k
  unsigned short* qb  = (unsigned short*)take(M * N * 2);
  unsigned short* kb  = (unsigned short*)take(M * N * 2);
  unsigned short* vb  = (unsigned short*)take(M * N * 2);
  unsigned short* ab  = (unsigned short*)take(M * N * 2);

  // convert inputs to bf16
  cvt_f32_bf16_kernel<<<4096, 256, 0, stream>>>(x, xb, M * D);
  cvt_f32_bf16_kernel<<<2048, 256, 0, stream>>>(Wq, wqb, (long)D * D);
  cvt_f32_bf16_kernel<<<2048, 256, 0, stream>>>(Wk, wkb, (long)D * D);
  cvt_f32_bf16_kernel<<<2048, 256, 0, stream>>>(Wv, wvb, (long)D * D);
  cvt_f32_bf16_kernel<<<2048, 256, 0, stream>>>(Wo, wob, (long)D * D);

  dim3 gg(M / 128, N / 128);   // 32 x 16

  // Q = x Wq^T -> rmsnorm+rope -> bf16
  gemm_nt_bf16_kernel<<<gg, 256, 0, stream>>>(xb, wqb, tf, nullptr, M, N, D, 0);
  rmsnorm_rope_kernel<<<(B * S * H_NUM) / 8, 256, 0, stream>>>(tf, qnw, rope, qb, S);
  // K = x Wk^T -> rmsnorm+rope -> bf16
  gemm_nt_bf16_kernel<<<gg, 256, 0, stream>>>(xb, wkb, tf, nullptr, M, N, D, 0);
  rmsnorm_rope_kernel<<<(B * S * H_NUM) / 8, 256, 0, stream>>>(tf, knw, rope, kb, S);
  // V = x Wv^T -> bf16 directly
  gemm_nt_bf16_kernel<<<gg, 256, 0, stream>>>(xb, wvb, nullptr, vb, M, N, D, 1);

  // flash attention per (b,h)
  attn_kernel<<<dim3(S / 128, B * H_NUM), 256, 0, stream>>>(qb, kb, vb, ab, S);

  // out = attn Wo^T (fp32 to d_out)
  gemm_nt_bf16_kernel<<<gg, 256, 0, stream>>>(ab, wob, (float*)d_out, nullptr, M, N, D, 0);
}
